// LinearAttention_24824910971194
// MI455X (gfx1250) — compile-verified, matrix operand order checked by reading, not run on hardware
//
#include <hip/hip_runtime.h>

// Linear attention: Y = (X Wq) cumstate-attn (X Wk, X Wv) then Y Wo.
// B=2, L=1024, D=1024, H=16, d=64.
// Matmuls: v_wmma_f32_16x16x32_bf16.  GEMM operand feed: TDM tensor_load_to_lds
// (double-buffered LDS tiles, s_wait_tensorcnt).

typedef __attribute__((ext_vector_type(16))) __bf16 v16bf;
typedef __attribute__((ext_vector_type(8)))  float  v8f;
typedef __attribute__((ext_vector_type(4)))  unsigned int v4u;
typedef __attribute__((ext_vector_type(8)))  int v8i;
typedef __attribute__((ext_vector_type(4)))  int v4i;

static __device__ __forceinline__ v8f wmma_bf16(v16bf a, v16bf b, v8f c) {
    return __builtin_amdgcn_wmma_f32_16x16x32_bf16(
        /*neg_a=*/false, a, /*neg_b=*/false, b,
        /*c_mod=*/(short)0, c, /*reuse_a=*/false, /*reuse_b=*/false);
}

// ---------------------------------------------------------------------------
// TDM: issue a 2D bf16 tile load Global -> LDS (D# per CDNA5 ISA ch.8).
// Rows of tile_d0 elements (64 -> 128B = 32 DWORDs) padded by 8 DWORDs (32B)
// => effective LDS row stride 80 bf16 elements, matching the [..][80] arrays.
// ---------------------------------------------------------------------------
static __device__ __forceinline__ void tdm_load_2d(
    unsigned lds_addr, const void* gaddr,
    unsigned tensor_d0, unsigned tensor_d1, unsigned stride_d0,
    unsigned tile_d0, unsigned tile_d1)
{
    unsigned long long ga = (unsigned long long)(size_t)gaddr;
    v4u g0;
    g0[0] = 1u;                                   // count=1, user descriptor
    g0[1] = lds_addr;                             // LDS byte address (bits 63:32)
    g0[2] = (unsigned)(ga & 0xFFFFFFFFu);         // global_addr[31:0]
    g0[3] = (unsigned)((ga >> 32) & 0x1FFFFFFu)   // global_addr[56:32]
          | (2u << 30);                           // type = 2 ("image")
    v8i g1;
    g1[0] = (int)((1u << 16)      // data_size code 1 = 2 bytes; workgroup_mask=0
                | (1u << 20)      // pad_enable (LDS dest padding)
                | (4u << 22)      // pad_interval: 32 DWORDs (one 128B row)
                | (7u << 25));    // pad_amount:   8 DWORDs (32B)
    g1[1] = (int)((tensor_d0 & 0xFFFFu) << 16);                 // tensor_dim0[15:0] @bit48
    g1[2] = (int)(((tensor_d0 >> 16) & 0xFFFFu)
                | ((tensor_d1 & 0xFFFFu) << 16));               // dim0 hi | dim1 lo
    g1[3] = (int)(((tensor_d1 >> 16) & 0xFFFFu)
                | ((tile_d0 & 0xFFFFu) << 16));                 // dim1 hi | tile_dim0
    g1[4] = (int)(tile_d1 & 0xFFFFu);                           // tile_dim1 (tile_dim2=0)
    g1[5] = (int)stride_d0;                                     // tensor_dim0_stride lo
    g1[6] = 0;                                                  // stride hi / dim1_stride
    g1[7] = 0;
    v4i z4 = {0, 0, 0, 0};                                      // 2D: groups 2/3 unused
    v8i z8 = {0, 0, 0, 0, 0, 0, 0, 0};                          // extra words (6-arg form)
    __builtin_amdgcn_tensor_load_to_lds(g0, g1, z4, z4, z8, /*cpol=*/0);
}

// ---------------------------------------------------------------------------
// fp32 -> bf16 elementwise convert (for X)
// ---------------------------------------------------------------------------
__global__ void __launch_bounds__(256)
cvt_f32_bf16(const float* __restrict__ in, __bf16* __restrict__ out, int n) {
    int i = blockIdx.x * blockDim.x + threadIdx.x;
    int stride = gridDim.x * blockDim.x;
    for (; i < n; i += stride) out[i] = (__bf16)in[i];
}

// ---------------------------------------------------------------------------
// Transpose fp32 [K x N] -> bf16 [N x K]  (weights -> B-operand layout)
// ---------------------------------------------------------------------------
__global__ void __launch_bounds__(256)
transpose_f32_to_bf16(const float* __restrict__ W, __bf16* __restrict__ Wt,
                      int K, int N) {
    __shared__ float tile[32][33];
    const int n0 = blockIdx.x * 32;
    const int k0 = blockIdx.y * 32;
    const int tx = threadIdx.x;       // 0..31
    const int ty = threadIdx.y;       // 0..7
#pragma unroll
    for (int yy = ty; yy < 32; yy += 8)
        tile[yy][tx] = W[(size_t)(k0 + yy) * N + (n0 + tx)];
    __syncthreads();
#pragma unroll
    for (int yy = ty; yy < 32; yy += 8)
        Wt[(size_t)(n0 + yy) * K + (k0 + tx)] = (__bf16)tile[tx][yy];
}

// ---------------------------------------------------------------------------
// GEMM: C[MxN] = A[MxK] * B, with B given transposed: Bt[NxK] row-major.
// Block: 256 thr = 8 waves, block tile 128(M) x 64(N), 32x32 per wave.
// Operands staged in LDS by the Tensor Data Mover, double-buffered (K-step 64).
// ---------------------------------------------------------------------------
template <typename OutT>
__global__ void __launch_bounds__(256)
gemm_bt_wmma(const __bf16* __restrict__ A, const __bf16* __restrict__ Bt,
             OutT* __restrict__ C, int M, int N, int K) {
    __shared__ __attribute__((aligned(32))) __bf16 As[2][128][80];  // 64 data + 16 pad
    __shared__ __attribute__((aligned(32))) __bf16 Bs[2][64][80];

    const int lane = threadIdx.x & 31;
    const int wave = threadIdx.x >> 5;     // 0..7
    const int half = lane >> 4;            // K sub-block selector
    const int r    = lane & 15;            // row within fragment
    const int wm   = wave >> 1;            // 0..3
    const int wn   = wave & 1;             // 0..1
    const int m_blk = blockIdx.y * 128;
    const int n_blk = blockIdx.x * 64;
    const int nk    = K / 64;

    if (threadIdx.x < 32) {                // wave 0 drives the TDM
        tdm_load_2d((unsigned)(size_t)&As[0][0][0], A  + (size_t)m_blk * K,
                    K, M, K, 64, 128);
        tdm_load_2d((unsigned)(size_t)&Bs[0][0][0], Bt + (size_t)n_blk * K,
                    K, N, K, 64, 64);
    }

    v8f acc[2][2] = {};
    for (int it = 0; it < nk; ++it) {
        const int cur = it & 1;
        if (threadIdx.x < 32) {
            if (it + 1 < nk) {             // prefetch next K-slab into other buffer
                const int k0 = (it + 1) * 64;
                tdm_load_2d((unsigned)(size_t)&As[1 - cur][0][0],
                            A  + (size_t)m_blk * K + k0, K, M, K, 64, 128);
                tdm_load_2d((unsigned)(size_t)&Bs[1 - cur][0][0],
                            Bt + (size_t)n_blk * K + k0, K, N, K, 64, 64);
                __builtin_amdgcn_s_wait_tensorcnt(2);  // current buffer landed
            } else {
                __builtin_amdgcn_s_wait_tensorcnt(0);
            }
        }
        __syncthreads();                   // publish LDS tiles to all waves

#pragma unroll
        for (int kk = 0; kk < 64; kk += 32) {
            v16bf a[2], b[2];
            a[0] = *(const v16bf*)(&As[cur][wm * 32 +      r][kk + half * 16]);
            a[1] = *(const v16bf*)(&As[cur][wm * 32 + 16 + r][kk + half * 16]);
            b[0] = *(const v16bf*)(&Bs[cur][wn * 32 +      r][kk + half * 16]);
            b[1] = *(const v16bf*)(&Bs[cur][wn * 32 + 16 + r][kk + half * 16]);
#pragma unroll
            for (int tm = 0; tm < 2; ++tm)
#pragma unroll
                for (int tn = 0; tn < 2; ++tn)
                    acc[tm][tn] = wmma_bf16(a[tm], b[tn], acc[tm][tn]);
        }
        __syncthreads();                   // done with 'cur' before it is refilled
    }

    // Epilogue: C/D layout -> lane<16: row=t, lane>=16: row=t+8; col = lane&15
#pragma unroll
    for (int tm = 0; tm < 2; ++tm)
#pragma unroll
        for (int tn = 0; tn < 2; ++tn)
#pragma unroll
            for (int t = 0; t < 8; ++t) {
                int row = m_blk + wm * 32 + 16 * tm + t + half * 8;
                int col = n_blk + wn * 32 + 16 * tn + r;
                C[(size_t)row * N + col] = (OutT)acc[tm][tn][t];
            }
}

// ---------------------------------------------------------------------------
// Chunked causal linear attention, one workgroup per (b,h); 4 waves.
// State kept as S^T (e,d) in fp32 LDS; per chunk (C=64):
//   Amask = causal(Q Kc^T);  Y = Q S_prev + Amask Vc;  S^T += Vc^T Kc
// ---------------------------------------------------------------------------
__global__ void __launch_bounds__(128)
linattn_kernel(const __bf16* __restrict__ Qb, const __bf16* __restrict__ Kb,
               const __bf16* __restrict__ Vb, __bf16* __restrict__ Yb) {
    const int bh   = blockIdx.x;       // 0..31
    const int bidx = bh >> 4;
    const int h    = bh & 15;
    const int tid  = threadIdx.x;
    const int wave = tid >> 5;         // 0..3 -> 16-row strip
    const int lane = tid & 31;
    const int half = lane >> 4;
    const int r    = lane & 15;

    __shared__ __attribute__((aligned(32))) float  St[64][64];  // S^T fp32
    __shared__ __attribute__((aligned(32))) __bf16 Sb[64][80];  // S^T bf16 snapshot
    __shared__ __attribute__((aligned(32))) __bf16 Am[64][80];  // masked QK^T
    __shared__ __attribute__((aligned(32))) __bf16 Kt[64][80];  // Kc^T [d][j]
    __shared__ __attribute__((aligned(32))) __bf16 Vt[64][80];  // Vc^T [e][j]

    for (int i = tid; i < 64 * 64; i += 128) St[i >> 6][i & 63] = 0.f;
    __syncthreads();

    const size_t headoff = (size_t)bidx * 1024 * 1024 + (size_t)h * 64;
    const __bf16* Qh = Qb + headoff;
    const __bf16* Kh = Kb + headoff;
    const __bf16* Vh = Vb + headoff;
    __bf16*       Yh = Yb + headoff;

    for (int c = 0; c < 16; ++c) {
        const size_t coff = (size_t)c * 64 * 1024;
        const __bf16* Qc = Qh + coff;
        const __bf16* Kc = Kh + coff;
        const __bf16* Vc = Vh + coff;

        // Stage K^T and V^T into LDS (coalesced global reads).
        for (int i = tid; i < 64 * 64; i += 128) {
            int j = i >> 6, x = i & 63;
            Kt[x][j] = Kc[(size_t)j * 1024 + x];
            Vt[x][j] = Vc[(size_t)j * 1024 + x];
        }

        // A = Qc Kc^T (row-major Kc already serves the B-operand layout).
        v8f a_acc[4] = {};
        for (int kk = 0; kk < 64; kk += 32) {
            v16bf aq = *(const v16bf*)(Qc + (size_t)(16 * wave + r) * 1024 + kk + half * 16);
#pragma unroll
            for (int tn = 0; tn < 4; ++tn) {
                v16bf bk = *(const v16bf*)(Kc + (size_t)(16 * tn + r) * 1024 + kk + half * 16);
                a_acc[tn] = wmma_bf16(aq, bk, a_acc[tn]);
            }
        }
        // Causal mask (diagonal kept: j <= i) and spill to LDS as bf16.
#pragma unroll
        for (int tn = 0; tn < 4; ++tn)
#pragma unroll
            for (int t = 0; t < 8; ++t) {
                int i = 16 * wave + t + half * 8;
                int j = 16 * tn + r;
                Am[i][j] = (__bf16)((i >= j) ? a_acc[tn][t] : 0.f);
            }
        // Snapshot state to bf16 for the WMMA B-operand.
        for (int i = tid; i < 64 * 64; i += 128)
            Sb[i >> 6][i & 63] = (__bf16)St[i >> 6][i & 63];
        __syncthreads();

        // Y strip: inter-chunk (Q * S_prev) + intra-chunk (Amask * Vc).
        v8f y_acc[4] = {};
        for (int kk = 0; kk < 64; kk += 32) {
            v16bf aq = *(const v16bf*)(Qc + (size_t)(16 * wave + r) * 1024 + kk + half * 16);
#pragma unroll
            for (int tn = 0; tn < 4; ++tn) {
                v16bf bs = *(const v16bf*)(&Sb[16 * tn + r][kk + half * 16]);
                y_acc[tn] = wmma_bf16(aq, bs, y_acc[tn]);
            }
        }
        for (int kk = 0; kk < 64; kk += 32) {
            v16bf aa = *(const v16bf*)(&Am[16 * wave + r][kk + half * 16]);
#pragma unroll
            for (int tn = 0; tn < 4; ++tn) {
                v16bf bv = *(const v16bf*)(&Vt[16 * tn + r][kk + half * 16]);
                y_acc[tn] = wmma_bf16(aa, bv, y_acc[tn]);
            }
        }
        __bf16* Yc = Yh + coff;
#pragma unroll
        for (int tn = 0; tn < 4; ++tn)
#pragma unroll
            for (int t = 0; t < 8; ++t) {
                int i = 16 * wave + t + half * 8;
                int j = 16 * tn + r;
                Yc[(size_t)i * 1024 + j] = (__bf16)y_acc[tn][t];
            }

        // State update: S^T[e][d] += sum_j Vt[e][j] * Kt[d][j]
        v8f s_acc[4];
#pragma unroll
        for (int tn = 0; tn < 4; ++tn)
#pragma unroll
            for (int t = 0; t < 8; ++t)
                s_acc[tn][t] = St[16 * wave + t + half * 8][16 * tn + r];
        for (int kk = 0; kk < 64; kk += 32) {
            v16bf av = *(const v16bf*)(&Vt[16 * wave + r][kk + half * 16]);
#pragma unroll
            for (int tn = 0; tn < 4; ++tn) {
                v16bf bk = *(const v16bf*)(&Kt[16 * tn + r][kk + half * 16]);
                s_acc[tn] = wmma_bf16(av, bk, s_acc[tn]);
            }
        }
#pragma unroll
        for (int tn = 0; tn < 4; ++tn)
#pragma unroll
            for (int t = 0; t < 8; ++t)
                St[16 * wave + t + half * 8][16 * tn + r] = s_acc[tn][t];

        __syncthreads();  // protects St/Kt/Vt for next chunk
    }
}

// ---------------------------------------------------------------------------
extern "C" void kernel_launch(void* const* d_in, const int* in_sizes, int n_in,
                              void* d_out, int out_size, void* d_ws, size_t ws_size,
                              hipStream_t stream) {
    constexpr int Bc = 2, Lc = 1024, Dc = 1024;
    constexpr int M = Bc * Lc;                 // 2048
    const float* X  = (const float*)d_in[0];
    const float* Wq = (const float*)d_in[1];
    const float* Wk = (const float*)d_in[2];
    const float* Wv = (const float*)d_in[3];
    const float* Wo = (const float*)d_in[4];
    float* Out = (float*)d_out;

    // Workspace carve-up (~28 MiB total, 256B-aligned slices).
    char* ws = (char*)d_ws;
    size_t off = 0;
    auto carve = [&](size_t bytes) -> void* {
        void* p = ws + off;
        off += (bytes + 255) & ~(size_t)255;
        return p;
    };
    __bf16* Xb  = (__bf16*)carve((size_t)M * Dc * 2);
    __bf16* Wqt = (__bf16*)carve((size_t)Dc * Dc * 2);
    __bf16* Wkt = (__bf16*)carve((size_t)Dc * Dc * 2);
    __bf16* Wvt = (__bf16*)carve((size_t)Dc * Dc * 2);
    __bf16* Wot = (__bf16*)carve((size_t)Dc * Dc * 2);
    __bf16* Qb  = (__bf16*)carve((size_t)M * Dc * 2);
    __bf16* Kbf = (__bf16*)carve((size_t)M * Dc * 2);
    __bf16* Vbf = (__bf16*)carve((size_t)M * Dc * 2);
    __bf16* Ybf = (__bf16*)carve((size_t)M * Dc * 2);

    // 1) Precision conversion + weight transposes.
    cvt_f32_bf16<<<512, 256, 0, stream>>>(X, Xb, M * Dc);
    dim3 tgrid(Dc / 32, Dc / 32), tblk(32, 8);
    transpose_f32_to_bf16<<<tgrid, tblk, 0, stream>>>(Wq, Wqt, Dc, Dc);
    transpose_f32_to_bf16<<<tgrid, tblk, 0, stream>>>(Wk, Wkt, Dc, Dc);
    transpose_f32_to_bf16<<<tgrid, tblk, 0, stream>>>(Wv, Wvt, Dc, Dc);
    transpose_f32_to_bf16<<<tgrid, tblk, 0, stream>>>(Wo, Wot, Dc, Dc);

    // 2) Q/K/V projections (TDM-fed WMMA GEMMs, bf16 out).
    dim3 ggrid(Dc / 64, M / 128), gblk(256);
    gemm_bt_wmma<__bf16><<<ggrid, gblk, 0, stream>>>(Xb, Wqt, Qb,  M, Dc, Dc);
    gemm_bt_wmma<__bf16><<<ggrid, gblk, 0, stream>>>(Xb, Wkt, Kbf, M, Dc, Dc);
    gemm_bt_wmma<__bf16><<<ggrid, gblk, 0, stream>>>(Xb, Wvt, Vbf, M, Dc, Dc);

    // 3) Chunked causal linear attention (one block per (b,h)).
    linattn_kernel<<<32, 128, 0, stream>>>(Qb, Kbf, Vbf, Ybf);

    // 4) Output projection (fp32 out).
    gemm_bt_wmma<float><<<ggrid, gblk, 0, stream>>>(Ybf, Wot, Out, M, Dc, Dc);
}
